// MultiViewTransformerLayer_49452253446272
// MI455X (gfx1250) — matrix operand order repaired and checked by reading, not compiled, not run on hardware
//
#include <hip/hip_runtime.h>
#include <hip/hip_bf16.h>
#include <math.h>

// ---------------- problem constants ----------------
enum : int {
  B_ = 8, V_ = 4, J_ = 512, C_ = 256, H_ = 64, W_ = 64,
  NH_ = 8, P_ = 16, HD_ = 32, FF_ = 1024, HW_ = H_ * W_
};

typedef __attribute__((ext_vector_type(16))) _Float16 v16h;
typedef __attribute__((ext_vector_type(8)))  float    v8f;

// ---------------- CDNA5 async global->LDS staging --------------------
#if __has_builtin(__builtin_amdgcn_global_load_async_to_lds_b128)
#define HAS_ASYNC_LDS 1
#endif

// exact parameter type per hipcc diagnostic:
//   int __attribute__((vector_size(16))) __device__ *  (AS1)
//   and the LDS analog in AS3.
typedef int ai4 __attribute__((vector_size(16)));

__device__ __forceinline__ void stage16B(const float* g, float* l) {
#ifdef HAS_ASYNC_LDS
  // GLOBAL_LOAD_ASYNC_TO_LDS_B128: 16B/lane straight to LDS, no VGPR round-trip
  __builtin_amdgcn_global_load_async_to_lds_b128(
      (__attribute__((address_space(1))) ai4*)g,
      (__attribute__((address_space(3))) ai4*)l, 0, 0);
#else
  l[0] = g[0]; l[1] = g[1]; l[2] = g[2]; l[3] = g[3];
#endif
}

__device__ __forceinline__ void wait_stage() {
#ifdef HAS_ASYNC_LDS
  asm volatile("s_wait_asynccnt 0x0" ::: "memory");
#endif
}

// =====================================================================
// WMMA GEMM: Y[M,N] = act(A[M,K] @ W[K,N] + bias[N])
// Block = 256 thr = 8 waves, tile 128x64, K-step 32 (== WMMA K).
// Waves arranged 4 (M) x 2 (N); each wave owns a 32x32 micro-tile:
// 2 A-frags x 2 B-frags -> 4 WMMA accumulators per K-step.
// Requires M % 128 == 0, N % 64 == 0, K % 32 == 0 (true for all calls).
// f32 tiles staged asynchronously to LDS; f32->f16 cvt on fragment build.
// =====================================================================
#define GBM 128
#define GBN 64
#define GBK 32

template <bool RELU>
__global__ void __launch_bounds__(256)
gemm_f16_wmma(const float* __restrict__ A, const float* __restrict__ Wm,
              const float* __restrict__ bias, float* __restrict__ Y,
              int M, int N, int K) {
  // strides padded to multiples of 4 floats: keeps every 16B chunk aligned
  __shared__ float Asf[GBM][GBK + 4];   // 128 x 36 f32 = 18 KB
  __shared__ float Bsf[GBK][GBN + 4];   //  32 x 68 f32 =  8.5 KB

  const int tid  = threadIdx.x;
  const int lane = tid & 31;
  const int wave = tid >> 5;
  const int wrow = wave >> 1;   // 0..3 -> rows wrow*32
  const int wcol = wave & 1;    // 0..1 -> cols wcol*32
  const int hi   = lane >> 4;   // half-wave select
  const int ln   = lane & 15;

  const int m0 = blockIdx.y * GBM;
  const int n0 = blockIdx.x * GBN;

  v8f acc00 = {}, acc01 = {}, acc10 = {}, acc11 = {};

  for (int k0 = 0; k0 < K; k0 += GBK) {
    // ---- stage A tile 128x32 f32: 1024 x 16B chunks, 4 per thread ----
#pragma unroll
    for (int c = 0; c < 4; ++c) {
      const int chunk = tid + c * 256;       // 0..1023
      const int r  = chunk >> 3;             // 0..127
      const int kk = (chunk & 7) * 4;        // 0..28
      const float* src = A + (size_t)(m0 + r) * K + k0 + kk;
      __builtin_prefetch(src + GBK, 0, 0);   // global_prefetch (next K tile)
      stage16B(src, &Asf[r][kk]);
    }
    // ---- stage B tile 32x64 f32: 512 x 16B chunks, 2 per thread ----
#pragma unroll
    for (int c = 0; c < 2; ++c) {
      const int chunk = tid + c * 256;       // 0..511
      const int r  = chunk >> 4;             // 0..31
      const int nn = (chunk & 15) * 4;       // 0..60
      const float* src = Wm + (size_t)(k0 + r) * N + n0 + nn;
      __builtin_prefetch(src + (size_t)GBK * N, 0, 0);
      stage16B(src, &Bsf[r][nn]);
    }
    wait_stage();
    __syncthreads();

    // ---- A fragments (16x32 each): element i -> K = hi*16 + i
    //      lanes 0-15: K 0-15 ; lanes 16-31: K 16-31
    v16h af0, af1;
    {
      const int r0 = wrow * 32 + ln;
      const int r1 = r0 + 16;
#pragma unroll
      for (int i = 0; i < 8; ++i) {
        af0[i]     = (_Float16)Asf[r0][hi * 16 + i];
        af0[i + 8] = (_Float16)Asf[r0][hi * 16 + 8 + i];
        af1[i]     = (_Float16)Asf[r1][hi * 16 + i];
        af1[i + 8] = (_Float16)Asf[r1][hi * 16 + 8 + i];
      }
    }
    // ---- B fragments (32x16 each): element i -> K = hi*16 + i, col = ln
    v16h bf0, bf1;
    {
      const int c0 = wcol * 32 + ln;
#pragma unroll
      for (int i = 0; i < 16; ++i) {
        bf0[i] = (_Float16)Bsf[hi * 16 + i][c0];
        bf1[i] = (_Float16)Bsf[hi * 16 + i][c0 + 16];
      }
    }
    acc00 = __builtin_amdgcn_wmma_f32_16x16x32_f16(false, af0, false, bf0,
                                                   (short)0, acc00, false, false);
    acc01 = __builtin_amdgcn_wmma_f32_16x16x32_f16(false, af0, false, bf1,
                                                   (short)0, acc01, false, false);
    acc10 = __builtin_amdgcn_wmma_f32_16x16x32_f16(false, af1, false, bf0,
                                                   (short)0, acc10, false, false);
    acc11 = __builtin_amdgcn_wmma_f32_16x16x32_f16(false, af1, false, bf1,
                                                   (short)0, acc11, false, false);
    __syncthreads();
  }

  // ---- store D: row = base + hi*8 + r, col = ln ----
  const int rowb = m0 + wrow * 32 + hi * 8;
  const int col0 = n0 + wcol * 32 + ln;
  const float b0 = bias[col0];
  const float b1 = bias[col0 + 16];
#pragma unroll
  for (int r = 0; r < 8; ++r) {
    float y00 = acc00[r] + b0, y01 = acc01[r] + b1;
    float y10 = acc10[r] + b0, y11 = acc11[r] + b1;
    if (RELU) {
      y00 = fmaxf(y00, 0.f); y01 = fmaxf(y01, 0.f);
      y10 = fmaxf(y10, 0.f); y11 = fmaxf(y11, 0.f);
    }
    Y[(size_t)(rowb + r) * N + col0]           = y00;
    Y[(size_t)(rowb + r) * N + col0 + 16]      = y01;
    Y[(size_t)(rowb + 16 + r) * N + col0]      = y10;
    Y[(size_t)(rowb + 16 + r) * N + col0 + 16] = y11;
  }
}

// =====================================================================
// Softmax over P=16 per (b,j,nh); in-place on (B*J, NH*P) buffer.
// =====================================================================
__global__ void softmax_p(float* __restrict__ a, int total) {
  int i = blockIdx.x * blockDim.x + threadIdx.x;
  if (i >= total) return;
  const int bj = i >> 3, nh = i & 7;
  float* p = a + (size_t)bj * (NH_ * P_) + nh * P_;
  float m = -1e30f;
#pragma unroll
  for (int k = 0; k < P_; ++k) m = fmaxf(m, p[k]);
  float e[P_], s = 0.f;
#pragma unroll
  for (int k = 0; k < P_; ++k) { e[k] = expf(p[k] - m); s += e[k]; }
  const float inv = 1.f / s;
#pragma unroll
  for (int k = 0; k < P_; ++k) p[k] = e[k] * inv;
}

// =====================================================================
// Deformable bilinear sampling + attention-weight reduction.
// value L2-resident (134MB < 192MB L2). Block = (b,v,j); thread = (nh,d).
// =====================================================================
__global__ void __launch_bounds__(256)
deform_sample(const float* __restrict__ value,  // (B*V, HW, C)
              const float* __restrict__ offs,   // (B*J, NH*P*2)
              const float* __restrict__ attw,   // (B*J, NH*P) softmaxed
              const float* __restrict__ anch,   // (B, V, J, 2)
              float* __restrict__ out) {        // (B*V*J, C)
  const int blk = blockIdx.x;          // (b*V+v)*J + j
  const int j  = blk % J_;
  const int bv = blk / J_;
  const int v  = bv % V_;
  const int b  = bv / V_;
  const int t  = threadIdx.x;
  const int nh = t >> 5;
  const int d  = t & 31;

  const size_t arow = ((size_t)(b * V_ + v) * J_ + j) * 2;
  const float ax = anch[arow + 0];
  const float ay = anch[arow + 1];
  const float* op = offs + (size_t)(b * J_ + j) * (NH_ * P_ * 2) + nh * (P_ * 2);
  const float* ap = attw + (size_t)(b * J_ + j) * (NH_ * P_)     + nh * P_;
  const float* vb = value + (size_t)bv * HW_ * C_ + nh * HD_ + d;

  float acc = 0.f;
#pragma unroll 4
  for (int p = 0; p < P_; ++p) {
    const float px = ax * (float)W_ + op[p * 2 + 0] - 0.5f;
    const float py = ay * (float)H_ + op[p * 2 + 1] - 0.5f;
    const float fx = floorf(px), fy = floorf(py);
    const float tx = px - fx, ty = py - fy;
    const int x0 = (int)fx, y0 = (int)fy;

    auto g = [&](int xi, int yi) -> float {
      if (xi < 0 || xi >= W_ || yi < 0 || yi >= H_) return 0.f;
      return vb[(size_t)(yi * W_ + xi) * C_];
    };
    const float s = (1.f - tx) * (1.f - ty) * g(x0, y0)
                  + tx * (1.f - ty)         * g(x0 + 1, y0)
                  + (1.f - tx) * ty         * g(x0, y0 + 1)
                  + tx * ty                 * g(x0 + 1, y0 + 1);
    acc += ap[p] * s;
  }
  out[(size_t)blk * C_ + nh * HD_ + d] = acc;
}

// =====================================================================
// Apply validity mask and permute (B,V,J,C) -> (B,J,V*C) for fuse GEMM.
// =====================================================================
__global__ void mask_permute(const float* __restrict__ din,
                             const int* __restrict__ valid,
                             float* __restrict__ dst) {
  const size_t i = (size_t)blockIdx.x * blockDim.x + threadIdx.x;
  if (i >= (size_t)B_ * J_ * V_ * C_) return;
  const int c = (int)(i & (C_ - 1));
  const size_t r = i >> 8;            // (b*J+j)*V + v
  const int v = (int)(r % V_);
  const size_t bj = r / V_;
  const int j = (int)(bj % J_);
  const int b = (int)(bj / J_);
  const size_t in_row = (size_t)(b * V_ + v) * J_ + j;
  const float m = valid[in_row] ? 1.f : 0.f;
  dst[i] = din[in_row * C_ + c] * m;
}

// =====================================================================
// Residual + LayerNorm over C=256; one 256-thread block per row.
// =====================================================================
__global__ void __launch_bounds__(256)
add_layernorm(const float* __restrict__ a, const float* __restrict__ bres,
              const float* __restrict__ g, const float* __restrict__ beta,
              float* __restrict__ out) {
  __shared__ float red[8];
  const int row = blockIdx.x;
  const int t = threadIdx.x;
  const size_t idx = (size_t)row * C_ + t;
  const float x = a[idx] + bres[idx];

  float s = x;
#pragma unroll
  for (int o = 16; o > 0; o >>= 1) s += __shfl_xor(s, o, 32);
  if ((t & 31) == 0) red[t >> 5] = s;
  __syncthreads();
  float tot = 0.f;
#pragma unroll
  for (int i = 0; i < 8; ++i) tot += red[i];
  const float mean = tot * (1.f / C_);
  __syncthreads();

  const float dlt = x - mean;
  float vs = dlt * dlt;
#pragma unroll
  for (int o = 16; o > 0; o >>= 1) vs += __shfl_xor(vs, o, 32);
  if ((t & 31) == 0) red[t >> 5] = vs;
  __syncthreads();
  float vtot = 0.f;
#pragma unroll
  for (int i = 0; i < 8; ++i) vtot += red[i];
  const float var = vtot * (1.f / C_);

  out[idx] = dlt * rsqrtf(var + 1e-5f) * g[t] + beta[t];
}

// =====================================================================
// Spatial self-attention per (b, head, 16-row query tile). HD=32 == WMMA K.
// One wave per block: QK^T via 32 WMMAs -> 16x512 scores in LDS -> softmax
// -> P@V via 16 K-chunks x 2 d-tiles of WMMA.
// =====================================================================
__global__ void __launch_bounds__(32)
spatial_attn(const float* __restrict__ Q, const float* __restrict__ K,
             const float* __restrict__ Vv, float* __restrict__ sp) {
  __shared__ float s[16][J_ + 4];

  const int blk = blockIdx.x;           // ((b*NH+nh) * 32) + jt
  const int jt  = blk & 31;             // J/16 = 32 tiles
  const int hb  = blk >> 5;
  const int nh  = hb & 7;
  const int b   = hb >> 3;
  const int lane = threadIdx.x;
  const int hi = lane >> 4, ln = lane & 15;
  const float scale = 0.1767766952966369f;   // 1/sqrt(HD)

  // Q fragment (16 x 32), pre-scaled
  v16h qa;
  {
    const int jq = jt * 16 + ln;
    const float* qp = Q + (size_t)(b * J_ + jq) * C_ + nh * HD_;
#pragma unroll
    for (int i = 0; i < 8; ++i) {
      qa[i]     = (_Float16)(qp[hi * 16 + i] * scale);
      qa[i + 8] = (_Float16)(qp[hi * 16 + 8 + i] * scale);
    }
  }
  // scores = Q @ K^T, one 16x16 tile per iteration
  for (int ct = 0; ct < J_ / 16; ++ct) {
    v16h kb;
    const int jk = ct * 16 + ln;
    const float* kp = K + (size_t)(b * J_ + jk) * C_ + nh * HD_;
#pragma unroll
    for (int i = 0; i < 16; ++i) kb[i] = (_Float16)kp[hi * 16 + i];
    v8f c = {};
    c = __builtin_amdgcn_wmma_f32_16x16x32_f16(false, qa, false, kb,
                                               (short)0, c, false, false);
#pragma unroll
    for (int r = 0; r < 8; ++r) s[hi * 8 + r][ct * 16 + ln] = c[r];
  }
  __syncthreads();

  // row softmax: lane pair (ln, ln+16) splits the 512 columns
  {
    const int r = ln;
    float m = -1e30f;
    for (int c2 = hi * 256; c2 < hi * 256 + 256; ++c2) m = fmaxf(m, s[r][c2]);
    m = fmaxf(m, __shfl_xor(m, 16, 32));
    float sum = 0.f;
    for (int c2 = hi * 256; c2 < hi * 256 + 256; ++c2) {
      const float e = expf(s[r][c2] - m);
      s[r][c2] = e;
      sum += e;
    }
    sum += __shfl_xor(sum, 16, 32);
    const float inv = 1.f / sum;
    for (int c2 = hi * 256; c2 < hi * 256 + 256; ++c2) s[r][c2] *= inv;
  }
  __syncthreads();

  // out = P @ V : accumulate over 16 chunks of 32 keys, 2 d-tiles
  v8f a0 = {}, a1 = {};
  for (int ct = 0; ct < J_ / 32; ++ct) {
    v16h pa;
#pragma unroll
    for (int i = 0; i < 8; ++i) {
      pa[i]     = (_Float16)s[ln][ct * 32 + hi * 16 + i];
      pa[i + 8] = (_Float16)s[ln][ct * 32 + hi * 16 + 8 + i];
    }
    v16h vb0, vb1;
#pragma unroll
    for (int i = 0; i < 16; ++i) {
      const int jk = ct * 32 + hi * 16 + i;
      const float* vp = Vv + (size_t)(b * J_ + jk) * C_ + nh * HD_;
      vb0[i] = (_Float16)vp[ln];
      vb1[i] = (_Float16)vp[16 + ln];
    }
    a0 = __builtin_amdgcn_wmma_f32_16x16x32_f16(false, pa, false, vb0,
                                                (short)0, a0, false, false);
    a1 = __builtin_amdgcn_wmma_f32_16x16x32_f16(false, pa, false, vb1,
                                                (short)0, a1, false, false);
  }
#pragma unroll
  for (int r = 0; r < 8; ++r) {
    const int jq = jt * 16 + hi * 8 + r;
    float* o = sp + (size_t)(b * J_ + jq) * C_ + nh * HD_;
    o[ln]      = a0[r];
    o[16 + ln] = a1[r];
  }
}

// =====================================================================
// Host-side orchestration
// =====================================================================
extern "C" void kernel_launch(void* const* d_in, const int* in_sizes, int n_in,
                              void* d_out, int out_size, void* d_ws, size_t ws_size,
                              hipStream_t stream) {
  (void)in_sizes; (void)n_in; (void)out_size; (void)ws_size;

  const float* feat_query = (const float*)d_in[0];   // (B,J,C)
  const float* frame_feat = (const float*)d_in[1];   // (B,V,HW,C)
  const float* anchors2d  = (const float*)d_in[2];   // (B,V,J,2)
  const int*   anchvalid  = (const int*)  d_in[3];   // (B,V,J)
  const float* off_w  = (const float*)d_in[4];
  const float* off_b  = (const float*)d_in[5];
  const float* aw_w   = (const float*)d_in[6];
  const float* aw_b   = (const float*)d_in[7];
  const float* val_w  = (const float*)d_in[8];
  const float* val_b  = (const float*)d_in[9];
  const float* dout_w = (const float*)d_in[10];
  const float* dout_b = (const float*)d_in[11];
  const float* fuse_w = (const float*)d_in[12];
  const float* fuse_b = (const float*)d_in[13];
  const float* ln_cg  = (const float*)d_in[14];
  const float* ln_cb  = (const float*)d_in[15];
  const float* ln_sg  = (const float*)d_in[16];
  const float* ln_sb  = (const float*)d_in[17];
  const float* ln_fg  = (const float*)d_in[18];
  const float* ln_fb  = (const float*)d_in[19];
  const float* q_w = (const float*)d_in[20];
  const float* q_b = (const float*)d_in[21];
  const float* k_w = (const float*)d_in[22];
  const float* k_b = (const float*)d_in[23];
  const float* v_w = (const float*)d_in[24];
  const float* v_b = (const float*)d_in[25];
  const float* o_w = (const float*)d_in[26];
  const float* o_b = (const float*)d_in[27];
  const float* w1 = (const float*)d_in[28];
  const float* b1 = (const float*)d_in[29];
  const float* w2 = (const float*)d_in[30];
  const float* b2 = (const float*)d_in[31];

  float* ws = (float*)d_ws;
  const size_t MB_f = 1048576;   // 1M floats
  float* value    = ws;                         // 32 Mf (134 MB, L2-resident)
  float* offsb    = ws + 32 * MB_f;             //  1 Mf (B*J, 256)
  float* attwb    = ws + 33 * MB_f;             //  0.5 Mf (B*J, 128)
  float* attn_out = ws + 34 * MB_f;             //  4 Mf (B*V*J, C)
  // buffers reusing the value region (value is dead after deform_sample)
  float* xb   = value + 0 * MB_f;
  float* Qb   = value + 1 * MB_f;
  float* Kb   = value + 2 * MB_f;
  float* Vb   = value + 3 * MB_f;
  float* spb  = value + 4 * MB_f;
  float* spo  = value + 5 * MB_f;
  float* x2b  = value + 6 * MB_f;
  float* ff1  = value + 7 * MB_f;               // 4 Mf (B*J, FF)
  float* ff2  = value + 11 * MB_f;
  float* fusein = value + 12 * MB_f;            // 4 Mf (B*J, V*C)
  float* fusedb = value + 16 * MB_f;            // 1 Mf
  float* doutb  = value + 17 * MB_f;            // 4 Mf (B*V*J, C)

  const dim3 blk256(256);

  // 1) value projection: (B*V*HW, C) @ (C, C)  -- the big streaming GEMM
  gemm_f16_wmma<false><<<dim3(C_ / GBN, (B_ * V_ * HW_) / GBM), blk256, 0, stream>>>(
      frame_feat, val_w, val_b, value, B_ * V_ * HW_, C_, C_);

  // 2) offsets + attention-weight logits from the query
  gemm_f16_wmma<false><<<dim3((NH_ * P_ * 2) / GBN, (B_ * J_) / GBM), blk256, 0, stream>>>(
      feat_query, off_w, off_b, offsb, B_ * J_, NH_ * P_ * 2, C_);
  gemm_f16_wmma<false><<<dim3((NH_ * P_) / GBN, (B_ * J_) / GBM), blk256, 0, stream>>>(
      feat_query, aw_w, aw_b, attwb, B_ * J_, NH_ * P_, C_);

  // 3) softmax over the P sampling points
  softmax_p<<<dim3((B_ * J_ * NH_ + 255) / 256), blk256, 0, stream>>>(attwb, B_ * J_ * NH_);

  // 4) bilinear gather + weighted reduce (value tensor stays in L2)
  deform_sample<<<dim3(B_ * V_ * J_), blk256, 0, stream>>>(
      value, offsb, attwb, anchors2d, attn_out);

  // 5) output projection of deformable attention
  gemm_f16_wmma<false><<<dim3(C_ / GBN, (B_ * V_ * J_) / GBM), blk256, 0, stream>>>(
      attn_out, dout_w, dout_b, doutb, B_ * V_ * J_, C_, C_);

  // 6) mask invalid anchors, permute (B,V,J,C) -> (B,J,V*C)
  mask_permute<<<dim3((B_ * J_ * V_ * C_) / 256), blk256, 0, stream>>>(
      doutb, anchvalid, fusein);

  // 7) multi-view fuse: (B*J, V*C) @ (V*C, C)
  gemm_f16_wmma<false><<<dim3(C_ / GBN, (B_ * J_) / GBM), blk256, 0, stream>>>(
      fusein, fuse_w, fuse_b, fusedb, B_ * J_, C_, V_ * C_);

  // 8) x = LN(q + fused)
  add_layernorm<<<dim3(B_ * J_), blk256, 0, stream>>>(feat_query, fusedb, ln_cg, ln_cb, xb);

  // 9) spatial self-attention QKV projections
  gemm_f16_wmma<false><<<dim3(C_ / GBN, (B_ * J_) / GBM), blk256, 0, stream>>>(
      xb, q_w, q_b, Qb, B_ * J_, C_, C_);
  gemm_f16_wmma<false><<<dim3(C_ / GBN, (B_ * J_) / GBM), blk256, 0, stream>>>(
      xb, k_w, k_b, Kb, B_ * J_, C_, C_);
  gemm_f16_wmma<false><<<dim3(C_ / GBN, (B_ * J_) / GBM), blk256, 0, stream>>>(
      xb, v_w, v_b, Vb, B_ * J_, C_, C_);

  // 10) per-head attention (WMMA QK^T + softmax + WMMA PV)
  spatial_attn<<<dim3(B_ * NH_ * (J_ / 16)), dim3(32), 0, stream>>>(Qb, Kb, Vb, spb);

  // 11) output projection + LN
  gemm_f16_wmma<false><<<dim3(C_ / GBN, (B_ * J_) / GBM), blk256, 0, stream>>>(
      spb, o_w, o_b, spo, B_ * J_, C_, C_);
  add_layernorm<<<dim3(B_ * J_), blk256, 0, stream>>>(xb, spo, ln_sg, ln_sb, x2b);

  // 12) FFN (ReLU fused in first GEMM epilogue) + final LN -> d_out
  gemm_f16_wmma<true><<<dim3(FF_ / GBN, (B_ * J_) / GBM), blk256, 0, stream>>>(
      x2b, w1, b1, ff1, B_ * J_, FF_, C_);
  gemm_f16_wmma<false><<<dim3(C_ / GBN, (B_ * J_) / GBM), blk256, 0, stream>>>(
      ff1, w2, b2, ff2, B_ * J_, C_, FF_);
  add_layernorm<<<dim3(B_ * J_), blk256, 0, stream>>>(x2b, ff2, ln_fg, ln_fb, (float*)d_out);
}
